// MHA_76553497084175
// MI455X (gfx1250) — matrix operand order repaired and checked by reading, not compiled, not run on hardware
//
#include <hip/hip_runtime.h>

// ---------------------------------------------------------------------------
// Causal MHA forward for MI455X (gfx1250, wave32, WMMA).
//   y = softmax(mask(QK^T)*hd^-0.5) V @ Wo^T + b,   also returns K, V [b,h,s,hd]
// All GEMMs + attention use v_wmma_f32_16x16x32_bf16 (bf16 in, fp32 accum).
// __launch_bounds__(256, 1) gives each wave the full VGPR budget so the
// 64x64 accumulator tiles (128 VGPRs) stay resident (no scratch spills).
// ---------------------------------------------------------------------------

typedef __bf16 bf16;
typedef __attribute__((ext_vector_type(16))) __bf16 v16bf;
typedef __attribute__((ext_vector_type(8)))  __bf16 v8bf;
typedef __attribute__((ext_vector_type(8)))  float  v8f;

#define BD 2
#define SD 2048
#define DD 1024
#define HN 16
#define HD 64

union U8F   { v8f  v; float e[8]; };
union U8BF  { v8bf v; bf16  e[8]; };
union U16BF { v16bf v; v8bf h[2]; };

static __device__ __forceinline__ v8f wmma_bf16(v16bf a, v16bf b, v8f c) {
  return __builtin_amdgcn_wmma_f32_16x16x32_bf16(false, a, false, b, (short)0, c,
                                                 false, false);
}

// A-fragment (16xK tile, row-major, K contiguous). p points at row start + k0.
// Lane layout: lanes 0-15 hold K 0..15; lanes 16-31 hold K 16..31.
static __device__ __forceinline__ v16bf load_afrag(const bf16* p, int half) {
  U16BF u;
  u.h[0] = *(const v8bf*)(p + 16 * half);
  u.h[1] = *(const v8bf*)(p + 16 * half + 8);
  return u.v;
}

// B-fragment (Kx16 tile; lane n holds column n, K contiguous in source row n).
// p points at "column row" start + k0. Lanes 0-15: K 0..15; 16-31: K 16..31.
static __device__ __forceinline__ v16bf load_bfrag(const bf16* p, int half) {
  return *(const v16bf*)(p + 16 * half);
}

// ---------------------------------------------------------------------------
// fp32 -> bf16 conversion
// ---------------------------------------------------------------------------
__global__ void cvt_f32_bf16(const float* __restrict__ src, bf16* __restrict__ dst,
                             int n) {
  int i = blockIdx.x * 256 + threadIdx.x;
  if (i < n) dst[i] = (bf16)src[i];
}

// ---------------------------------------------------------------------------
// QKV projection: C[4096,3072] = Xbf[4096,1024] * Wbf[3072,1024]^T
// One wave = 64x64 tile (16 accumulators). Epilogue routes Q/K/V.
// ---------------------------------------------------------------------------
__global__ void __launch_bounds__(256, 1)
qkv_gemm(const bf16* __restrict__ xbf, const bf16* __restrict__ wbf,
         bf16* __restrict__ qbf, bf16* __restrict__ kbf,
         bf16* __restrict__ vT, float* __restrict__ k_out,
         float* __restrict__ v_out) {
  const int lane = threadIdx.x & 31;
  const int wave = threadIdx.x >> 5;
  const int half = lane >> 4;
  const int nid  = lane & 15;
  const int wt   = blockIdx.x * 8 + wave;   // 3072 wave tiles = (4096/64)*(3072/64)
  const int tn   = wt % 48, tm = wt / 48;
  const int m0   = tm * 64, n0 = tn * 64;

  v8f acc[16];
#pragma unroll
  for (int i = 0; i < 16; ++i) acc[i] = (v8f){};

  for (int k0 = 0; k0 < DD; k0 += 32) {
    v16bf a[4], b[4];
#pragma unroll
    for (int i = 0; i < 4; ++i)
      a[i] = load_afrag(xbf + (size_t)(m0 + i * 16 + nid) * DD + k0, half);
#pragma unroll
    for (int j = 0; j < 4; ++j)
      b[j] = load_bfrag(wbf + (size_t)(n0 + j * 16 + nid) * DD + k0, half);
#pragma unroll
    for (int i = 0; i < 4; ++i)
#pragma unroll
      for (int j = 0; j < 4; ++j) acc[i * 4 + j] = wmma_bf16(a[i], b[j], acc[i * 4 + j]);
  }

#pragma unroll
  for (int i = 0; i < 4; ++i) {
#pragma unroll
    for (int j = 0; j < 4; ++j) {
      U8F u; u.v = acc[i * 4 + j];
      const int n = n0 + j * 16 + nid;           // feature column (uniform region/tile)
      const int mbase = m0 + i * 16 + 8 * half;  // first of 8 consecutive rows
      if (n < DD) {                              // ---- Q (bf16 only) ----
#pragma unroll
        for (int rr = 0; rr < 8; ++rr)
          qbf[(size_t)(mbase + rr) * DD + n] = (bf16)u.e[rr];
      } else if (n < 2 * DD) {                   // ---- K (bf16 + fp32 out) ----
        const int nn = n - DD, hh = nn >> 6, e = nn & 63;
#pragma unroll
        for (int rr = 0; rr < 8; ++rr) {
          const int m = mbase + rr, bb = m >> 11, s = m & (SD - 1);
          kbf[(size_t)m * DD + nn] = (bf16)u.e[rr];
          k_out[((size_t)((bb * HN + hh) * SD + s)) * HD + e] = u.e[rr];
        }
      } else {                                   // ---- V (fp32 out + bf16 V^T) ----
        const int nn = n - 2 * DD, hh = nn >> 6, e = nn & 63;
        U8BF pk;
#pragma unroll
        for (int rr = 0; rr < 8; ++rr) {
          const int m = mbase + rr, bb = m >> 11, s = m & (SD - 1);
          v_out[((size_t)((bb * HN + hh) * SD + s)) * HD + e] = u.e[rr];
          pk.e[rr] = (bf16)u.e[rr];
        }
        const int bb = mbase >> 11, s0 = mbase & (SD - 1);
        *(v8bf*)(vT + ((size_t)((bb * HN + hh) * HD) + e) * SD + s0) = pk.v;
      }
    }
  }
}

// ---------------------------------------------------------------------------
// Flash-style causal attention. One wave = one 16-row q tile; 32 keys / iter.
// Computes S^T = K*Q^T (so per-query softmax stats are in-lane over VGPRs +
// one shfl_xor(16)), then O^T += V^T * P^T with P staged through LDS.
// ---------------------------------------------------------------------------
__global__ void __launch_bounds__(256, 1)
attn_kernel(const bf16* __restrict__ qbf, const bf16* __restrict__ kbf,
            const bf16* __restrict__ vT, bf16* __restrict__ attn) {
  __shared__ __align__(32) bf16 ldsP[8 * 512];  // 16(q) x 32(k) bf16 tile per wave
  const int lane = threadIdx.x & 31;
  const int wave = threadIdx.x >> 5;
  const int half = lane >> 4;
  const int nid  = lane & 15;

  const int tile = blockIdx.x * 8 + wave;  // 4096 tiles = B*H*(S/16)
  const int qt = tile & 127;
  const int bh = tile >> 7;
  const int b = bh >> 4, h = bh & 15;
  const int q0 = qt * 16;

  const bf16* Qb = qbf + (size_t)(b * SD) * DD + h * HD;
  const bf16* Kb = kbf + (size_t)(b * SD) * DD + h * HD;
  const bf16* Vb = vT + (size_t)bh * HD * SD;

  // Q^T B-fragments for e in [0,32) and [32,64): lane n reads Q row q0+n.
  const v16bf bq0 = *(const v16bf*)(Qb + (size_t)(q0 + nid) * DD + 16 * half);
  const v16bf bq1 = *(const v16bf*)(Qb + (size_t)(q0 + nid) * DD + 32 + 16 * half);

  v8f o[4];
#pragma unroll
  for (int t = 0; t < 4; ++t) o[t] = (v8f){};
  float m_run = -__builtin_inff();
  float l_run = 0.f;
  const float cs = 0.125f * 1.44269504088896340736f;  // hd^-0.5 * log2(e)
  const int qg = q0 + nid;
  bf16* lp = ldsP + wave * 512 + nid * 32;

  for (int k0 = 0; k0 < q0 + 16; k0 += 32) {
    // S^T tiles: keys [k0,k0+16) and [k0+16,k0+32) vs queries [q0,q0+16)
    v8f s0v = (v8f){}, s1v = (v8f){};
#pragma unroll
    for (int es = 0; es < 2; ++es) {
      const v16bf ak0 = load_afrag(Kb + (size_t)(k0 + nid) * DD + es * 32, half);
      const v16bf ak1 = load_afrag(Kb + (size_t)(k0 + 16 + nid) * DD + es * 32, half);
      const v16bf bq = es ? bq1 : bq0;
      s0v = wmma_bf16(ak0, bq, s0v);
      s1v = wmma_bf16(ak1, bq, s1v);
    }
    U8F s0; s0.v = s0v;
    U8F s1; s1.v = s1v;

    // causal mask + tile max (C layout: N=query=lane&15, M=key=rr+8*half)
    float mt = -__builtin_inff();
#pragma unroll
    for (int rr = 0; rr < 8; ++rr) {
      const int kg0 = k0 + 8 * half + rr;
      if (kg0 > qg)      s0.e[rr] = -__builtin_inff();
      if (kg0 + 16 > qg) s1.e[rr] = -__builtin_inff();
      mt = fmaxf(mt, fmaxf(s0.e[rr], s1.e[rr]));
    }
    mt = fmaxf(mt, __shfl_xor(mt, 16, 32));
    const float mnew  = fmaxf(m_run, mt);
    const float alpha = exp2f((m_run - mnew) * cs);

    float lt = 0.f;
#pragma unroll
    for (int rr = 0; rr < 8; ++rr) {
      const float p0 = exp2f((s0.e[rr] - mnew) * cs);
      const float p1 = exp2f((s1.e[rr] - mnew) * cs);
      s0.e[rr] = p0; s1.e[rr] = p1;
      lt += p0 + p1;
    }
    lt += __shfl_xor(lt, 16, 32);
    l_run = l_run * alpha + lt;
    m_run = mnew;

#pragma unroll
    for (int t = 0; t < 4; ++t) {
      U8F u; u.v = o[t];
#pragma unroll
      for (int rr = 0; rr < 8; ++rr) u.e[rr] *= alpha;
      o[t] = u.v;
    }

    // stage P^T (as P[q][k], 16x32 bf16) in LDS, reload as B-fragments
    U8BF pk0, pk1;
#pragma unroll
    for (int rr = 0; rr < 8; ++rr) {
      pk0.e[rr] = (bf16)s0.e[rr];
      pk1.e[rr] = (bf16)s1.e[rr];
    }
    *(v8bf*)(lp + 8 * half) = pk0.v;
    *(v8bf*)(lp + 16 + 8 * half) = pk1.v;
    asm volatile("s_wait_dscnt 0" ::: "memory");
    const v16bf bp = *(const v16bf*)(lp + 16 * half);

    // O^T[e,q] += V^T[e,k] * P^T[k,q]; V^T rows are k-contiguous
#pragma unroll
    for (int t = 0; t < 4; ++t) {
      const v16bf av = load_afrag(Vb + (size_t)(t * 16 + nid) * SD + k0, half);
      o[t] = wmma_bf16(av, bp, o[t]);
    }
  }

  // finalize: divide by l, store bf16 attn output (row s = q0+nid, cols h*64+e)
  const float inv_l = 1.0f / l_run;
  bf16* Ob = attn + ((size_t)(b * SD) + q0 + nid) * DD + h * HD;
#pragma unroll
  for (int t = 0; t < 4; ++t) {
    U8F u; u.v = o[t];
    U8BF pk;
#pragma unroll
    for (int rr = 0; rr < 8; ++rr) pk.e[rr] = (bf16)(u.e[rr] * inv_l);
    *(v8bf*)(Ob + t * 16 + 8 * half) = pk.v;
  }
}

// ---------------------------------------------------------------------------
// Output projection: Y[4096,1024] = attn_bf[4096,1024] * Wo[1024,1024]^T + b
// ---------------------------------------------------------------------------
__global__ void __launch_bounds__(256, 1)
oproj_gemm(const bf16* __restrict__ attn, const bf16* __restrict__ wobf,
           const float* __restrict__ bias, float* __restrict__ y) {
  const int lane = threadIdx.x & 31;
  const int wave = threadIdx.x >> 5;
  const int half = lane >> 4;
  const int nid  = lane & 15;
  const int wt   = blockIdx.x * 8 + wave;  // 1024 wave tiles = (4096/64)*(1024/64)
  const int tn   = wt % 16, tm = wt / 16;
  const int m0   = tm * 64, n0 = tn * 64;

  v8f acc[16];
#pragma unroll
  for (int i = 0; i < 16; ++i) acc[i] = (v8f){};

  for (int k0 = 0; k0 < DD; k0 += 32) {
    v16bf a[4], b[4];
#pragma unroll
    for (int i = 0; i < 4; ++i)
      a[i] = load_afrag(attn + (size_t)(m0 + i * 16 + nid) * DD + k0, half);
#pragma unroll
    for (int j = 0; j < 4; ++j)
      b[j] = load_bfrag(wobf + (size_t)(n0 + j * 16 + nid) * DD + k0, half);
#pragma unroll
    for (int i = 0; i < 4; ++i)
#pragma unroll
      for (int j = 0; j < 4; ++j) acc[i * 4 + j] = wmma_bf16(a[i], b[j], acc[i * 4 + j]);
  }

#pragma unroll
  for (int i = 0; i < 4; ++i) {
#pragma unroll
    for (int j = 0; j < 4; ++j) {
      U8F u; u.v = acc[i * 4 + j];
      const int n = n0 + j * 16 + nid;
      const int mbase = m0 + i * 16 + 8 * half;
      const float bv = bias[n];
#pragma unroll
      for (int rr = 0; rr < 8; ++rr)
        y[(size_t)(mbase + rr) * DD + n] = u.e[rr] + bv;
    }
  }
}

// ---------------------------------------------------------------------------
// Host launcher. Workspace layout (48 MiB total):
//   [0,8Mi) x_bf16 | [8,14Mi) wqkv_bf16 | [14,16Mi) wo_bf16 | [16,24Mi) q_bf16
//   [24,32Mi) k_bf16 | [32,40Mi) vT_bf16 [b,h,e,s] | [40,48Mi) attn_bf16
// ---------------------------------------------------------------------------
extern "C" void kernel_launch(void* const* d_in, const int* in_sizes, int n_in,
                              void* d_out, int out_size, void* d_ws, size_t ws_size,
                              hipStream_t stream) {
  const float* x    = (const float*)d_in[0];
  const float* wqkv = (const float*)d_in[1];
  const float* wo   = (const float*)d_in[2];
  const float* wob  = (const float*)d_in[3];

  float* y     = (float*)d_out;
  float* k_out = y + (size_t)BD * SD * DD;
  float* v_out = k_out + (size_t)BD * SD * DD;

  char* ws = (char*)d_ws;
  bf16* xbf  = (bf16*)(ws);
  bf16* wqbf = (bf16*)(ws + ((size_t)8 << 20));
  bf16* wobf = (bf16*)(ws + ((size_t)14 << 20));
  bf16* qbf  = (bf16*)(ws + ((size_t)16 << 20));
  bf16* kbf  = (bf16*)(ws + ((size_t)24 << 20));
  bf16* vT   = (bf16*)(ws + ((size_t)32 << 20));
  bf16* attn = (bf16*)(ws + ((size_t)40 << 20));

  const int nx = BD * SD * DD;            // 4,194,304
  const int nw = 3 * DD * DD;             // 3,145,728
  const int no = DD * DD;                 // 1,048,576
  cvt_f32_bf16<<<nx / 256, 256, 0, stream>>>(x, xbf, nx);
  cvt_f32_bf16<<<nw / 256, 256, 0, stream>>>(wqkv, wqbf, nw);
  cvt_f32_bf16<<<no / 256, 256, 0, stream>>>(wo, wobf, no);

  qkv_gemm<<<384, 256, 0, stream>>>(xbf, wqbf, qbf, kbf, vT, k_out, v_out);
  attn_kernel<<<512, 256, 0, stream>>>(qbf, kbf, vT, attn);
  oproj_gemm<<<128, 256, 0, stream>>>(attn, wobf, wob, y);
}
